// ParallelMambaVIX_39771397161703
// MI455X (gfx1250) — compile-verified, matrix operand order checked by reading, not run on hardware
//
#include <hip/hip_runtime.h>
#include <hip/hip_bf16.h>
#include <math.h>

typedef __attribute__((ext_vector_type(16))) _Float16 v16h;
typedef __attribute__((ext_vector_type(8)))  float    v8f;
typedef __attribute__((ext_vector_type(4)))  unsigned int u32x4;
typedef __attribute__((ext_vector_type(4)))  int      i32x4;
typedef __attribute__((ext_vector_type(8)))  int      i32x8;

#define BB 2
#define TT 2048
#define DM 256
#define DI 512
#define DS 16
#define DR 16
#define KC 4
#define HID 128
#define NH 4

__device__ __forceinline__ float gelu_f(float x) {
    // tanh approximation (jax.nn.gelu default approximate=True)
    float x3 = x * x * x;
    return 0.5f * x * (1.0f + tanhf(0.7978845608028654f * (x + 0.044715f * x3)));
}
__device__ __forceinline__ float silu_f(float x) {
    return x / (1.0f + __expf(-x));
}

// ---------------------------------------------------------------------------
// Generic WMMA GEMM: C[M,N] = act(A[M,K](lda) * W[K,N](ldw) + bias) (+C if addC)
// One wave computes a 16 x (NT*16) output tile.
//  - A tile: loaded DIRECTLY from global per-lane as 4x float4 (each lane owns
//    the two 8-float row segments its WMMA A-layout slot needs; no LDS).
//  - B tile (NT==2): staged to LDS by the Tensor Data Mover (tensor_load_to_lds)
//    with hardware row padding (32 DWORD rows + 8 DWORD pad -> Bs[32][40]) and
//    hardware zero-fill of rows beyond K (tensor_dim1 = K-kk).
//  - B tile (NT==3): float4 staging loop (pad interval must be a power of two).
// Requires: lda,ldw % 4 == 0, K % 16 == 0, M % 16 == 0, N % (NT*16) == 0.
// ---------------------------------------------------------------------------
template<int NT>
__global__ __launch_bounds__(32)
void wmma_gemm_kernel(const float* __restrict__ A, int lda,
                      const float* __restrict__ W, int ldw,
                      const float* __restrict__ bias,
                      float* __restrict__ C, int ldc,
                      int M, int N, int K, int addC, int act)
{
    constexpr int BW   = NT * 16;      // B tile width
    constexpr int BSTR = BW + 8;       // padded LDS row stride
    __shared__ float Bs[32][BSTR];
    const int lane = threadIdx.x;
    const int n0 = blockIdx.x * BW;
    const int m0 = blockIdx.y * 16;
    const int mr = lane & 15;             // A row / B col for this lane
    const int kh = (lane >> 4) << 3;      // lanes 0-15: K 0..7/16..23; 16-31: 8..15/24..31

    v8f zero = {};
    v8f acc[NT];
    #pragma unroll
    for (int q = 0; q < NT; q++) acc[q] = zero;

    const float* arow = A + (size_t)(m0 + mr) * lda;

    for (int kk = 0; kk < K; kk += 32) {
        if (kk + 32 < K) {
            // speculative prefetch of the next K-tile (global_prefetch_b8)
            __builtin_prefetch(arow + kk + 32, 0, 0);
            __builtin_prefetch(W + (size_t)(kk + 32 + mr) * ldw + n0, 0, 0);
        }

        if constexpr (NT == 2) {
            // ---- B tile via Tensor Data Mover: 2D tile, 4B elements ----
            unsigned ldsb = (unsigned)(uintptr_t)(void*)&Bs[0][0];
            unsigned long long ga =
                (unsigned long long)(uintptr_t)(W + (size_t)kk * ldw + n0);
            int rows = K - kk; if (rows > 32) rows = 32;   // zero-fill beyond
            // D# group 0: count=1 | lds_addr | global_addr | type=2
            u32x4 g0 = { 1u, ldsb, (unsigned)ga,
                         ((unsigned)(ga >> 32) & 0x01FFFFFFu) | 0x80000000u };
            // D# group 1:
            //  w0: data_size=4B (2<<16) | pad_enable (1<<20) |
            //      pad_interval=32 DWORDs (4<<22) | pad_amount=8 DWORDs (7<<25)
            i32x8 g1;
            g1[0] = (int)((2u << 16) | (1u << 20) | (4u << 22) | (7u << 25));
            g1[1] = (int)((unsigned)BW << 16);      // tensor_dim0[15:0] @ bits 63:48
            g1[2] = (int)((unsigned)rows << 16);    // tensor_dim1[15:0] @ bits 95:80
            g1[3] = (int)((unsigned)BW << 16);      // tile_dim0        @ bits 127:112
            g1[4] = 32;                             // tile_dim1        @ bits 143:128
            g1[5] = ldw;                            // tensor_dim0_stride[31:0]
            g1[6] = 0;
            g1[7] = 0;
            i32x4 z4 = { 0, 0, 0, 0 };
#if defined(__clang_major__) && (__clang_major__ >= 23)
            i32x8 z8 = { 0, 0, 0, 0, 0, 0, 0, 0 };
            __builtin_amdgcn_tensor_load_to_lds(g0, g1, z4, z4, z8, 0);
#else
            __builtin_amdgcn_tensor_load_to_lds(g0, g1, z4, z4, 0);
#endif
            __builtin_amdgcn_s_wait_tensorcnt((short)0);
            // compiler-level fence: TDM wrote LDS behind the compiler's back
            asm volatile("" ::: "memory");
        } else {
            // ---- B tile via float4 staging (NT*128 chunks, NT*4 per lane) ----
            #pragma unroll
            for (int i = 0; i < NT * 4; i++) {
                int idx = i * 32 + lane;
                int r   = idx / (NT * 4);
                int c4  = (idx % (NT * 4)) * 4;
                float4 v = make_float4(0.f, 0.f, 0.f, 0.f);
                if (kk + r < K)
                    v = *(const float4*)(W + (size_t)(kk + r) * ldw + n0 + c4);
                *(float4*)(&Bs[r][c4]) = v;
            }
        }
        __syncthreads();

        // ---- A fragment: direct global float4 loads, no LDS ----
        float4 f0 = *(const float4*)(arow + kk + kh);
        float4 f1 = *(const float4*)(arow + kk + kh + 4);
        float4 f2, f3;
        if (kk + 16 < K) {
            f2 = *(const float4*)(arow + kk + 16 + kh);
            f3 = *(const float4*)(arow + kk + 16 + kh + 4);
        } else {
            f2 = make_float4(0.f, 0.f, 0.f, 0.f);
            f3 = f2;
        }
        v16h a;
        a[0]  = (_Float16)f0.x;  a[1]  = (_Float16)f0.y;
        a[2]  = (_Float16)f0.z;  a[3]  = (_Float16)f0.w;
        a[4]  = (_Float16)f1.x;  a[5]  = (_Float16)f1.y;
        a[6]  = (_Float16)f1.z;  a[7]  = (_Float16)f1.w;
        a[8]  = (_Float16)f2.x;  a[9]  = (_Float16)f2.y;
        a[10] = (_Float16)f2.z;  a[11] = (_Float16)f2.w;
        a[12] = (_Float16)f3.x;  a[13] = (_Float16)f3.y;
        a[14] = (_Float16)f3.z;  a[15] = (_Float16)f3.w;

        #pragma unroll
        for (int q = 0; q < NT; q++) {
            v16h b;
            #pragma unroll
            for (int j = 0; j < 8; j++) {
                b[j]     = (_Float16)Bs[kh + j][q * 16 + mr];
                b[j + 8] = (_Float16)Bs[16 + kh + j][q * 16 + mr];
            }
            acc[q] = __builtin_amdgcn_wmma_f32_16x16x32_f16(
                         false, a, false, b, (short)0, acc[q], false, false);
        }
        __syncthreads();
    }

    // C/D layout: lane&15 = N column; VGPR r -> M = r (+8 for lanes 16-31)
    const int mb = m0 + ((lane >> 4) << 3);
    #pragma unroll
    for (int q = 0; q < NT; q++) {
        int ncol = n0 + q * 16 + (lane & 15);
        #pragma unroll
        for (int r = 0; r < 8; r++) {
            int row = mb + r;
            float v = acc[q][r];
            if (bias) v += bias[ncol];
            if (act == 3) v = (v > 20.0f) ? v : log1pf(__expf(v));   // softplus
            if (addC) v += C[(size_t)row * ldc + ncol];
            C[(size_t)row * ldc + ncol] = v;
        }
    }
}

// ---------------------------------------------------------------------------
// FiLM conditioning: m = gelu(LN(macro) @ W1 + b1) @ W2 + b2   (B x 256)
// ---------------------------------------------------------------------------
__global__ void film_kernel(const float* __restrict__ mc,
                            const float* __restrict__ flw, const float* __restrict__ flb,
                            const float* __restrict__ W1,  const float* __restrict__ b1,
                            const float* __restrict__ W2,  const float* __restrict__ b2,
                            float* __restrict__ m)
{
    __shared__ float sg[DM];
    int tid = threadIdx.x;
    for (int b = 0; b < BB; b++) {
        float mu = 0.0f;
        for (int i = 0; i < 64; i++) mu += mc[b * 64 + i];
        mu *= (1.0f / 64.0f);
        float var = 0.0f;
        for (int i = 0; i < 64; i++) { float d = mc[b * 64 + i] - mu; var += d * d; }
        var *= (1.0f / 64.0f);
        float inv = rsqrtf(var + 1e-5f);
        float acc = b1[tid];
        for (int i = 0; i < 64; i++) {
            float ln = (mc[b * 64 + i] - mu) * inv * flw[i] + flb[i];
            acc += ln * W1[i * DM + tid];
        }
        sg[tid] = gelu_f(acc);
        __syncthreads();
        float acc2 = b2[tid];
        for (int i = 0; i < DM; i++) acc2 += sg[i] * W2[i * DM + tid];
        m[b * DM + tid] = acc2;
        __syncthreads();
    }
}

// ---------------------------------------------------------------------------
// combined[b,t,:] = m[b,:] + time_encoding(ts[b,t])
// ---------------------------------------------------------------------------
__global__ void combined_kernel(const int* __restrict__ ts,
                                const float* __restrict__ m,
                                float* __restrict__ combined)
{
    int t = blockIdx.x, b = blockIdx.y, d = threadIdx.x;
    int tsv = ts[b * TT + t];
    float tod = (float)(tsv % 86400);
    const float cf = -9.210340371976184f / 128.0f;   // -ln(10000)/half
    float te;
    if (d < 128) te = __sinf(tod * __expf((float)d * cf));
    else         te = __cosf(tod * __expf((float)(d - 128) * cf));
    combined[((long)b * TT + t) * DM + d] = m[b * DM + d] + te;
}

// ---------------------------------------------------------------------------
// hn = gamma * LN(h) + beta     (gamma=gb[:,:256], beta=gb[:,256:])
// ---------------------------------------------------------------------------
__global__ void lnfilm_kernel(const float* __restrict__ h, const float* __restrict__ gb,
                              const float* __restrict__ lnw, const float* __restrict__ lnb,
                              float* __restrict__ hn)
{
    __shared__ float red[DM];
    int row = blockIdx.x, tid = threadIdx.x;
    float v = h[(long)row * DM + tid];
    red[tid] = v; __syncthreads();
    for (int s = 128; s > 0; s >>= 1) { if (tid < s) red[tid] += red[tid + s]; __syncthreads(); }
    float mu = red[0] * (1.0f / DM);
    __syncthreads();
    float dv = v - mu;
    red[tid] = dv * dv; __syncthreads();
    for (int s = 128; s > 0; s >>= 1) { if (tid < s) red[tid] += red[tid + s]; __syncthreads(); }
    float var = red[0] * (1.0f / DM);
    __syncthreads();
    float x = dv * rsqrtf(var + 1e-5f) * lnw[tid] + lnb[tid];
    float gamma = gb[(long)row * (2 * DM) + tid];
    float beta  = gb[(long)row * (2 * DM) + DM + tid];
    hn[(long)row * DM + tid] = gamma * x + beta;
}

// ---------------------------------------------------------------------------
// xconv[b,t,d] = silu(conv_b[d] + sum_k xc[b,t-3+k,d] * conv_w[d,k]); xc=xz[:,:,:512]
// ---------------------------------------------------------------------------
__global__ void conv_kernel(const float* __restrict__ xz,
                            const float* __restrict__ cw, const float* __restrict__ cb,
                            float* __restrict__ xconv)
{
    int t = blockIdx.x, b = blockIdx.y, d = threadIdx.x;
    float acc = cb[d];
    #pragma unroll
    for (int k = 0; k < KC; k++) {
        int tt = t - (KC - 1) + k;
        if (tt >= 0) acc += xz[((long)b * TT + tt) * (2 * DI) + d] * cw[d * KC + k];
    }
    xconv[((long)b * TT + t) * DI + d] = silu_f(acc);
}

// ---------------------------------------------------------------------------
// Selective scan: one channel per thread, 16 states in registers.
// B_t (16) and C_t (16) are 32 contiguous floats in proj -> staged to LDS with
// one wave via global_load_async_to_lds_b32 (ASYNCcnt path), then published by
// the workgroup barrier. Output fused with +x*D and *silu(z).
// ---------------------------------------------------------------------------
__global__ __launch_bounds__(256)
void scan_kernel(const float* __restrict__ delta, const float* __restrict__ xconv,
                 const float* __restrict__ proj,  const float* __restrict__ xz,
                 const float* __restrict__ Alog,  const float* __restrict__ Dp,
                 float* __restrict__ yg)
{
    __shared__ float sBC[32];     // [0..15] = B_t, [16..31] = C_t
    int tid = threadIdx.x;
    int b = blockIdx.x >> 1;
    int d = ((blockIdx.x & 1) << 8) + tid;

    float st[DS], Ad[DS];
    #pragma unroll
    for (int s = 0; s < DS; s++) { st[s] = 0.0f; Ad[s] = -__expf(Alog[d * DS + s]); }
    float Dd = Dp[d];

    // LDS byte offset of sBC: low 32 bits of the generic (flat-aperture) address
    unsigned ldsbase = (unsigned)(uintptr_t)(void*)sBC;

    for (int t = 0; t < TT; t++) {
        long row = (long)b * TT + t;
        if (tid < 32) {
            unsigned ldsoff = ldsbase + (unsigned)tid * 4u;
            const float* gsrc = proj + row * 48 + DR + tid;   // B_t then C_t, contiguous
            asm volatile("global_load_async_to_lds_b32 %0, %1, off"
                         :: "v"(ldsoff), "v"(gsrc) : "memory");
            asm volatile("s_wait_asynccnt 0x0" ::: "memory");
        }
        __syncthreads();
        float dt = delta[row * DI + d];
        float xt = xconv[row * DI + d];
        float dx = dt * xt;
        float y = 0.0f;
        #pragma unroll
        for (int s = 0; s < DS; s++) {
            st[s] = __expf(dt * Ad[s]) * st[s] + dx * sBC[s];
            y += st[s] * sBC[16 + s];
        }
        float zv = xz[row * (2 * DI) + DI + d];
        yg[row * DI + d] = (y + xt * Dd) * silu_f(zv);
        __syncthreads();
    }
}

// ---------------------------------------------------------------------------
// Trunk head: pooled -> gelu(LN(.@W1+b1)) -> gelu(LN(.@W2+b2)) -> @hor_W.T + hor_b
// ---------------------------------------------------------------------------
__device__ void ln_gelu_128(float* v, const float* w, const float* bb, float* red, int tid)
{
    if (tid < HID) red[tid] = v[tid];
    __syncthreads();
    for (int s = 64; s > 0; s >>= 1) { if (tid < s) red[tid] += red[tid + s]; __syncthreads(); }
    float mu = red[0] * (1.0f / HID);
    __syncthreads();
    if (tid < HID) { float d = v[tid] - mu; red[tid] = d * d; }
    __syncthreads();
    for (int s = 64; s > 0; s >>= 1) { if (tid < s) red[tid] += red[tid + s]; __syncthreads(); }
    float var = red[0] * (1.0f / HID);
    __syncthreads();
    if (tid < HID) {
        float x = (v[tid] - mu) * rsqrtf(var + 1e-5f) * w[tid] + bb[tid];
        v[tid] = gelu_f(x);
    }
    __syncthreads();
}

__global__ void tail_kernel(const float* __restrict__ h,
                            const float* __restrict__ W1, const float* __restrict__ b1,
                            const float* __restrict__ l1w, const float* __restrict__ l1b,
                            const float* __restrict__ W2, const float* __restrict__ b2,
                            const float* __restrict__ l2w, const float* __restrict__ l2b,
                            const float* __restrict__ hw, const float* __restrict__ hb,
                            float* __restrict__ out)
{
    __shared__ float sp[DM], t1[HID], t2[HID], red[HID];
    int tid = threadIdx.x;
    for (int b = 0; b < BB; b++) {
        sp[tid] = h[((long)b * TT + (TT - 1)) * DM + tid];
        __syncthreads();
        if (tid < HID) {
            float acc = b1[tid];
            for (int i = 0; i < DM; i++) acc += sp[i] * W1[i * HID + tid];
            t1[tid] = acc;
        }
        __syncthreads();
        ln_gelu_128(t1, l1w, l1b, red, tid);
        if (tid < HID) {
            float acc = b2[tid];
            for (int i = 0; i < HID; i++) acc += t1[i] * W2[i * HID + tid];
            t2[tid] = acc;
        }
        __syncthreads();
        ln_gelu_128(t2, l2w, l2b, red, tid);
        if (tid < NH) {
            float acc = hb[tid];
            for (int j = 0; j < HID; j++) acc += t2[j] * hw[tid * HID + j];
            out[b * NH + tid] = acc;
        }
        __syncthreads();
    }
}

// ---------------------------------------------------------------------------
extern "C" void kernel_launch(void* const* d_in, const int* in_sizes, int n_in,
                              void* d_out, int out_size, void* d_ws, size_t ws_size,
                              hipStream_t stream)
{
    (void)in_sizes; (void)n_in; (void)out_size; (void)ws_size;
    auto P = [&](int i) { return (const float*)d_in[i]; };
    const float* x     = P(0);
    const float* macro = P(1);
    const int*   ts    = (const int*)d_in[2];
    // params (recursive dict insertion order): 3..10 top-level, 11+13*L per layer, 63.. trunk

    const int M = BB * TT;   // 4096 rows
    float* ws = (float*)d_ws;
    size_t off = 0;
    auto alloc = [&](size_t n) { float* p = ws + off; off += n; return p; };
    float* h        = alloc((size_t)M * DM);
    float* combined = alloc((size_t)M * DM);
    float* hn       = alloc((size_t)M * DM);
    float* gb       = alloc((size_t)M * 2 * DM);   // reused as gated-y buffer
    float* xz       = alloc((size_t)M * 2 * DI);
    float* xconv    = alloc((size_t)M * DI);
    float* proj     = alloc((size_t)M * 48);
    float* delta    = alloc((size_t)M * DI);
    float* mbuf     = alloc(BB * DM);

    auto gemm = [&](const float* A, int lda, const float* W, int ldw,
                    const float* bias, float* C, int ldc, int N, int K,
                    int addC, int act) {
        if (N % 32 == 0)
            wmma_gemm_kernel<2><<<dim3(N / 32, M / 16), 32, 0, stream>>>(
                A, lda, W, ldw, bias, C, ldc, M, N, K, addC, act);
        else
            wmma_gemm_kernel<3><<<dim3(N / 48, M / 16), 32, 0, stream>>>(
                A, lda, W, ldw, bias, C, ldc, M, N, K, addC, act);
    };

    // FiLM + time-encoding conditioning
    film_kernel<<<1, DM, 0, stream>>>(macro, P(5), P(6), P(7), P(8), P(9), P(10), mbuf);
    combined_kernel<<<dim3(TT, BB), DM, 0, stream>>>(ts, mbuf, combined);

    // input embedding: h = x @ in_embed_W + b  (K=32)
    gemm(x, 32, P(3), DM, P(4), h, DM, DM, 32, 0, 0);

    for (int L = 0; L < 4; L++) {
        int base = 11 + 13 * L;
        const float* ln_w   = P(base + 0);
        const float* ln_b   = P(base + 1);
        const float* head_W = P(base + 2);
        const float* head_b = P(base + 3);
        const float* in_W   = P(base + 4);
        const float* conv_w = P(base + 5);
        const float* conv_b = P(base + 6);
        const float* xprojW = P(base + 7);
        const float* dt_W   = P(base + 8);
        const float* dt_b   = P(base + 9);
        const float* A_log  = P(base + 10);
        const float* Dp     = P(base + 11);
        const float* out_W  = P(base + 12);

        gemm(combined, DM, head_W, 2 * DM, head_b, gb, 2 * DM, 2 * DM, DM, 0, 0);
        lnfilm_kernel<<<M, DM, 0, stream>>>(h, gb, ln_w, ln_b, hn);
        gemm(hn, DM, in_W, 2 * DI, nullptr, xz, 2 * DI, 2 * DI, DM, 0, 0);
        conv_kernel<<<dim3(TT, BB), DI, 0, stream>>>(xz, conv_w, conv_b, xconv);
        gemm(xconv, DI, xprojW, 48, nullptr, proj, 48, 48, DI, 0, 0);
        gemm(proj, 48, dt_W, DI, dt_b, delta, DI, DI, DR, 0, 3);        // softplus
        scan_kernel<<<BB * (DI / 256), 256, 0, stream>>>(delta, xconv, proj, xz,
                                                         A_log, Dp, gb /*yg*/);
        gemm(gb, DI, out_W, DM, nullptr, h, DM, DM, DI, 1, 0);          // residual add
    }

    tail_kernel<<<1, DM, 0, stream>>>(h, P(63), P(64), P(65), P(66), P(67),
                                      P(68), P(69), P(70), P(71), P(72),
                                      (float*)d_out);
}
